// HBlock_22239340659138
// MI455X (gfx1250) — matrix operand order repaired and checked by reading, not compiled, not run on hardware
//
#include <hip/hip_runtime.h>

// ---------------------------------------------------------------------------
// CDNA5 (gfx1250) hyperbolic transformer block.
// All GEMMs run on v_wmma_f32_16x16x32_bf16 with an fp32 -> bf16 hi/lo split
// (3-product accumulation ~ fp32 accuracy, 2.7x the fp32-WMMA FLOP ceiling).
// GEMM tiles are staged with GLOBAL_LOAD_ASYNC_TO_LDS (ASYNCcnt) into a
// double-buffered LDS pipeline so WMMA on tile k overlaps the copy of k+1.
// B=8, L=1024, D=512, H=8, dh=64, MLP=2048.
// ---------------------------------------------------------------------------

typedef __attribute__((ext_vector_type(16))) __bf16 v16bf;
typedef __attribute__((ext_vector_type(8)))  float  v8f;
typedef int v4i_vs __attribute__((vector_size(16)));   // matches builtin param

union FragBF {
    v16bf v;
    unsigned u[8];
};

__device__ __forceinline__ unsigned short f2bf(float f) {
    unsigned u = __float_as_uint(f);
    u += 0x7FFFu + ((u >> 16) & 1u);      // round-to-nearest-even
    return (unsigned short)(u >> 16);
}
__device__ __forceinline__ float bf2f(unsigned short h) {
    return __uint_as_float(((unsigned)h) << 16);
}

__device__ __forceinline__ v8f wmma_bf(const FragBF& a, const FragBF& b, v8f c) {
    return __builtin_amdgcn_wmma_f32_16x16x32_bf16(
        /*neg_a=*/false, a.v, /*neg_b=*/false, b.v,
        /*c_mod=*/(short)0, c, /*reuse_a=*/false, /*reuse_b=*/false);
}

__device__ __forceinline__ float wreduce32(float v) {
    #pragma unroll
    for (int off = 16; off > 0; off >>= 1) v += __shfl_xor(v, off, 32);
    return v;
}

#define EPSF 1e-6f

// ---- gfx1250 async global->LDS copy (ASYNCcnt path), guarded ---------------
#if defined(__has_builtin)
#if __has_builtin(__builtin_amdgcn_global_load_async_to_lds_b128)
#define USE_ASYNC_LDS 1
#endif
#endif
#ifndef USE_ASYNC_LDS
#define USE_ASYNC_LDS 0
#endif

#if USE_ASYNC_LDS
__device__ __forceinline__ void async_cp128(const void* g, void* l) {
    __builtin_amdgcn_global_load_async_to_lds_b128(
        (__attribute__((address_space(1))) v4i_vs*)(g),
        (__attribute__((address_space(3))) v4i_vs*)(l), 0, 0);
}
#if __has_builtin(__builtin_amdgcn_s_wait_asynccnt)
#define WAIT_ASYNC(n) __builtin_amdgcn_s_wait_asynccnt(n)
#else
#define WAIT_ASYNC(n) asm volatile("s_wait_asynccnt %0" ::"i"(n) : "memory")
#endif
#endif

// ---------------------------------------------------------------------------
// fp32 -> bf16 hi/lo split (elementwise)
// ---------------------------------------------------------------------------
__global__ void k_split(const float* __restrict__ src,
                        unsigned short* __restrict__ hi,
                        unsigned short* __restrict__ lo, size_t n) {
    size_t i = (size_t)blockIdx.x * blockDim.x + threadIdx.x;
    if (i < n) {
        float f = src[i];
        unsigned short h = f2bf(f);
        hi[i] = h;
        lo[i] = f2bf(f - bf2f(h));
    }
}

// strip the time column off x (B,L,513) -> spatial (B,L,512) as hi/lo
__global__ void k_split_x(const float* __restrict__ x,
                          unsigned short* __restrict__ hi,
                          unsigned short* __restrict__ lo,
                          int rows, int D) {
    size_t i = (size_t)blockIdx.x * blockDim.x + threadIdx.x;
    size_t n = (size_t)rows * D;
    if (i < n) {
        size_t r = i / D, d = i % D;
        float f = x[r * (D + 1) + 1 + d];
        unsigned short h = f2bf(f);
        hi[i] = h;
        lo[i] = f2bf(f - bf2f(h));
    }
}

// ---------------------------------------------------------------------------
// Tiled WMMA GEMM:  out[m,n] = act( sum_k A[m,k] * W[n,k] + bias[n] )
// A: M x K (hi/lo bf16), W: N x K (hi/lo bf16). Tile 128x64x32, 256 thr / 8 waves,
// wave grid 4x2, each wave owns a 32x32 C block (2x2 of 16x16 WMMA tiles).
// bf16x2 3-product accumulation: hi*hi + hi*lo + lo*hi.
// Double-buffered LDS; tile k+1 streamed via async global->LDS while tile k
// is consumed by the WMMA pipes.
// ---------------------------------------------------------------------------
#define BM 128
#define BN 64
#define BK 32

template <int ACT>
__global__ __launch_bounds__(256) void gemm_bf16x2(
    const unsigned short* __restrict__ Ahi, const unsigned short* __restrict__ Alo,
    const unsigned short* __restrict__ Whi, const unsigned short* __restrict__ Wlo,
    const float* __restrict__ bias,
    float* __restrict__ outF, unsigned short* __restrict__ outHi,
    unsigned short* __restrict__ outLo, int M, int N, int K) {
    __shared__ unsigned short lAhi[2][BM * BK], lAlo[2][BM * BK];
    __shared__ unsigned short lWhi[2][BN * BK], lWlo[2][BN * BK];

    const int tid  = threadIdx.x;
    const int lane = tid & 31, wave = tid >> 5;
    const int wm = wave & 3, wn = wave >> 2;
    const int m0 = blockIdx.y * BM, n0 = blockIdx.x * BN;
    const int lm = lane & 15, lh = lane >> 4;

    // cooperative-load coordinates
    const int ar = tid >> 1, ac = (tid & 1) << 4;  // A: 16 bf16 each
    const int wr = tid >> 2, wc = (tid & 3) << 3;  // W:  8 bf16 each

    // stage one 32-wide k-tile into LDS buffer `buf`
    auto stage = [&](int buf, int k0) {
        const size_t ga = (size_t)(m0 + ar) * K + k0 + ac;
        const size_t gw = (size_t)(n0 + wr) * K + k0 + wc;
#if USE_ASYNC_LDS
        async_cp128(Ahi + ga,     &lAhi[buf][ar * BK + ac]);
        async_cp128(Ahi + ga + 8, &lAhi[buf][ar * BK + ac + 8]);
        async_cp128(Alo + ga,     &lAlo[buf][ar * BK + ac]);
        async_cp128(Alo + ga + 8, &lAlo[buf][ar * BK + ac + 8]);
        async_cp128(Whi + gw,     &lWhi[buf][wr * BK + wc]);
        async_cp128(Wlo + gw,     &lWlo[buf][wr * BK + wc]);
#else
        uint4 h0 = *(const uint4*)(Ahi + ga);
        uint4 h1 = *(const uint4*)(Ahi + ga + 8);
        uint4 l0 = *(const uint4*)(Alo + ga);
        uint4 l1 = *(const uint4*)(Alo + ga + 8);
        *(uint4*)&lAhi[buf][ar * BK + ac]     = h0;
        *(uint4*)&lAhi[buf][ar * BK + ac + 8] = h1;
        *(uint4*)&lAlo[buf][ar * BK + ac]     = l0;
        *(uint4*)&lAlo[buf][ar * BK + ac + 8] = l1;
        *(uint4*)&lWhi[buf][wr * BK + wc] = *(const uint4*)(Whi + gw);
        *(uint4*)&lWlo[buf][wr * BK + wc] = *(const uint4*)(Wlo + gw);
#endif
        __builtin_prefetch(Ahi + ga + BK, 0, 1);
        __builtin_prefetch(Whi + gw + BK, 0, 1);
    };

    v8f acc[2][2] = {};

    stage(0, 0);  // prologue: tile 0 in flight

    for (int k0 = 0; k0 < K; k0 += BK) {
        const int  cur  = (k0 / BK) & 1;
        const bool more = (k0 + BK) < K;
        if (more) stage(cur ^ 1, k0 + BK);  // stream next tile into other buffer
#if USE_ASYNC_LDS
        // wait until only the 6 just-issued copies remain => buf[cur] has landed
        if (more) WAIT_ASYNC(6); else WAIT_ASYNC(0);
#endif
        __syncthreads();

        // ---- build fragments from LDS ----
        FragBF aH[2], aL[2], bH[2], bL[2];
        #pragma unroll
        for (int mt = 0; mt < 2; ++mt) {
            int row = wm * 32 + mt * 16 + lm;
            #pragma unroll
            for (int i = 0; i < 8; ++i) {
                int kb = ((i < 4) ? 0 : 16) + (lh << 3) + 2 * (i & 3);  // A layout
                aH[mt].u[i] = *(const unsigned*)&lAhi[cur][row * BK + kb];
                aL[mt].u[i] = *(const unsigned*)&lAlo[cur][row * BK + kb];
            }
        }
        #pragma unroll
        for (int nt = 0; nt < 2; ++nt) {
            int rowN = wn * 32 + nt * 16 + lm;
            #pragma unroll
            for (int i = 0; i < 8; ++i) {
                int kb = ((i < 4) ? 0 : 16) + (lh << 3) + 2 * (i & 3);  // B layout
                bH[nt].u[i] = *(const unsigned*)&lWhi[cur][rowN * BK + kb];
                bL[nt].u[i] = *(const unsigned*)&lWlo[cur][rowN * BK + kb];
            }
        }

        // ---- 12 WMMAs per k-step ----
        #pragma unroll
        for (int mt = 0; mt < 2; ++mt)
            #pragma unroll
            for (int nt = 0; nt < 2; ++nt) {
                acc[mt][nt] = wmma_bf(aH[mt], bH[nt], acc[mt][nt]);
                acc[mt][nt] = wmma_bf(aH[mt], bL[nt], acc[mt][nt]);
                acc[mt][nt] = wmma_bf(aL[mt], bH[nt], acc[mt][nt]);
            }
        __syncthreads();  // everyone done reading buf[cur] before it is rewritten
    }

    // ---- epilogue: bias, activation, fp32 and/or hi-lo re-split ----
    #pragma unroll
    for (int mt = 0; mt < 2; ++mt)
        #pragma unroll
        for (int nt = 0; nt < 2; ++nt)
            #pragma unroll
            for (int j = 0; j < 8; ++j) {
                int row = m0 + wm * 32 + mt * 16 + (lh << 3) + j;
                int col = n0 + wn * 32 + nt * 16 + lm;
                float v = acc[mt][nt][j] + bias[col];
                if (ACT == 1) v = v > 0.f ? v : 0.f;
                size_t idx = (size_t)row * N + col;
                if (outF) outF[idx] = v;
                if (outHi) {
                    unsigned short h = f2bf(v);
                    outHi[idx] = h;
                    outLo[idx] = f2bf(v - bf2f(h));
                }
            }
}

// ---------------------------------------------------------------------------
// Hyperbolic attention. One block = 128 thr (4 waves) = 64 q-rows of one (b,h).
// KV streamed in 32-key tiles through LDS (shared by all 4 waves).
// scores <= 0 on the hyperboloid => streaming exp, no online max.
// renorm is scale-invariant => row-sum only needed for the EPS clip.
// ---------------------------------------------------------------------------
__global__ __launch_bounds__(128) void hattn_kernel(
    const unsigned short* __restrict__ qHi, const unsigned short* __restrict__ qLo,
    const unsigned short* __restrict__ kHi, const unsigned short* __restrict__ kLo,
    const unsigned short* __restrict__ vHi, const unsigned short* __restrict__ vLo,
    unsigned short* __restrict__ aggHi, unsigned short* __restrict__ aggLo,
    int B, int L, int H, int dh) {
    const int D = H * dh;  // 512
    const int qtiles = L / 64;
    const int bid = blockIdx.x;
    const int b  = bid / (H * qtiles);
    const int h  = (bid / qtiles) % H;
    const int q0 = (bid % qtiles) * 64;

    __shared__ unsigned short lKhi[32 * 64], lKlo[32 * 64];
    __shared__ unsigned short lVt[64 * 32];     // V transposed [dh][key]
    __shared__ unsigned short lP[4 * 16 * 32];  // per-wave P tile
    __shared__ float qt_s[64], ktacc[32], vtacc[32], kt_s[32], vt_s[32];

    const int lane = threadIdx.x & 31, wave = threadIdx.x >> 5;
    const int lm = lane & 15, lh = lane >> 4;

    // ---- Q fragments (two 32-wide k-steps over dh=64), straight from global ----
    FragBF qH[2], qL[2];
    {
        size_t rowBase = ((size_t)b * L + q0 + wave * 16 + lm) * D + h * dh;
        #pragma unroll
        for (int ks = 0; ks < 2; ++ks)
            #pragma unroll
            for (int i = 0; i < 8; ++i) {
                int kb = ks * 32 + ((i < 4) ? 0 : 16) + (lh << 3) + 2 * (i & 3);
                qH[ks].u[i] = *(const unsigned*)(qHi + rowBase + kb);
                qL[ks].u[i] = *(const unsigned*)(qLo + rowBase + kb);
            }
    }
    // qt = sqrt(1 + |q|^2): one thread per q-row
    if (threadIdx.x < 64) {
        size_t base = ((size_t)b * L + q0 + threadIdx.x) * D + h * dh;
        float ss = 0.f;
        for (int d = 0; d < 64; ++d) {
            float f = bf2f(qHi[base + d]) + bf2f(qLo[base + d]);
            ss += f * f;
        }
        qt_s[threadIdx.x] = sqrtf(1.f + ss);
    }
    __syncthreads();

    v8f   O[4] = {};
    float rs[8] = {}, aT[8] = {};

    for (int kt0 = 0; kt0 < L; kt0 += 32) {
        if (threadIdx.x < 32) { ktacc[threadIdx.x] = 0.f; vtacc[threadIdx.x] = 0.f; }
        __syncthreads();

        // ---- cooperative KV tile load: thread t -> key r = t/4, 16 dh elems ----
        {
            const int t = threadIdx.x;
            const int r = t >> 2, c0 = (t & 3) << 4;
            size_t g = ((size_t)b * L + kt0 + r) * D + h * dh + c0;

            uint4 kh0 = *(const uint4*)(kHi + g), kh1 = *(const uint4*)(kHi + g + 8);
            uint4 kl0 = *(const uint4*)(kLo + g), kl1 = *(const uint4*)(kLo + g + 8);
            *(uint4*)&lKhi[r * 64 + c0]     = kh0;
            *(uint4*)&lKhi[r * 64 + c0 + 8] = kh1;
            *(uint4*)&lKlo[r * 64 + c0]     = kl0;
            *(uint4*)&lKlo[r * 64 + c0 + 8] = kl1;
            unsigned khw[8] = {kh0.x, kh0.y, kh0.z, kh0.w, kh1.x, kh1.y, kh1.z, kh1.w};
            unsigned klw[8] = {kl0.x, kl0.y, kl0.z, kl0.w, kl1.x, kl1.y, kl1.z, kl1.w};
            float pk = 0.f;
            #pragma unroll
            for (int e = 0; e < 8; ++e) {
                float f0 = bf2f((unsigned short)(khw[e] & 0xffff)) +
                           bf2f((unsigned short)(klw[e] & 0xffff));
                float f1 = bf2f((unsigned short)(khw[e] >> 16)) +
                           bf2f((unsigned short)(klw[e] >> 16));
                pk += f0 * f0 + f1 * f1;
            }
            atomicAdd(&ktacc[r], pk);  // ds_add_f32

            uint4 vh0 = *(const uint4*)(vHi + g), vh1 = *(const uint4*)(vHi + g + 8);
            uint4 vl0 = *(const uint4*)(vLo + g), vl1 = *(const uint4*)(vLo + g + 8);
            unsigned vhw[8] = {vh0.x, vh0.y, vh0.z, vh0.w, vh1.x, vh1.y, vh1.z, vh1.w};
            unsigned vlw[8] = {vl0.x, vl0.y, vl0.z, vl0.w, vl1.x, vl1.y, vl1.z, vl1.w};
            float pv = 0.f;
            #pragma unroll
            for (int e = 0; e < 8; ++e) {
                unsigned short h0 = (unsigned short)(vhw[e] & 0xffff);
                unsigned short h1 = (unsigned short)(vhw[e] >> 16);
                lVt[(c0 + 2 * e) * 32 + r]     = h0;  // transpose into LDS
                lVt[(c0 + 2 * e + 1) * 32 + r] = h1;
                float f0 = bf2f(h0) + bf2f((unsigned short)(vlw[e] & 0xffff));
                float f1 = bf2f(h1) + bf2f((unsigned short)(vlw[e] >> 16));
                pv += f0 * f0 + f1 * f1;
            }
            atomicAdd(&vtacc[r], pv);
        }
        __syncthreads();
        if (threadIdx.x < 32) {
            kt_s[threadIdx.x] = sqrtf(1.f + ktacc[threadIdx.x]);
            vt_s[threadIdx.x] = sqrtf(1.f + vtacc[threadIdx.x]);
        }
        __syncthreads();

        // ---- per wave: S = Q Kt  (two 16-key subtiles -> one 16x32 P tile) ----
        #pragma unroll
        for (int sub = 0; sub < 2; ++sub) {
            FragBF bH[2], bL[2];
            int key = sub * 16 + lm;
            #pragma unroll
            for (int ks = 0; ks < 2; ++ks)
                #pragma unroll
                for (int i = 0; i < 8; ++i) {
                    int kb = ks * 32 + ((i < 4) ? 0 : 16) + (lh << 3) + 2 * (i & 3);
                    bH[ks].u[i] = *(const unsigned*)&lKhi[key * 64 + kb];
                    bL[ks].u[i] = *(const unsigned*)&lKlo[key * 64 + kb];
                }
            v8f S = {};
            #pragma unroll
            for (int ks = 0; ks < 2; ++ks) {
                S = wmma_bf(qH[ks], bH[ks], S);
                S = wmma_bf(qH[ks], bL[ks], S);
                S = wmma_bf(qL[ks], bH[ks], S);
            }
            // softmax weights (unnormalized; scores <= 0 so exp <= 1)
            float ktv = kt_s[sub * 16 + lm];
            float vtv = vt_s[sub * 16 + lm];
            int qrl = lh << 3;
            #pragma unroll
            for (int j = 0; j < 8; ++j) {
                float inner = S[j] - qt_s[wave * 16 + qrl + j] * ktv;
                float p = __expf(0.25f + 0.25f * inner);
                rs[j] += p;
                aT[j] += p * vtv;
                lP[wave * 512 + (qrl + j) * 32 + sub * 16 + lm] = f2bf(p);
            }
        }

        // ---- P @ V : P re-read from (per-wave) LDS in A-fragment layout ----
        FragBF pF;
        #pragma unroll
        for (int i = 0; i < 8; ++i) {
            int kb = ((i < 4) ? 0 : 16) + (lh << 3) + 2 * (i & 3);
            pF.u[i] = *(const unsigned*)&lP[wave * 512 + lm * 32 + kb];
        }
        #pragma unroll
        for (int nt = 0; nt < 4; ++nt) {
            FragBF bV;
            int dcol = nt * 16 + lm;
            #pragma unroll
            for (int i = 0; i < 8; ++i) {
                int kb = ((i < 4) ? 0 : 16) + (lh << 3) + 2 * (i & 3);
                bV.u[i] = *(const unsigned*)&lVt[dcol * 32 + kb];
            }
            O[nt] = wmma_bf(pF, bV, O[nt]);
        }
        __syncthreads();
    }

    // ---- epilogue: butterfly reductions over the 16-lane half, renorm ----
    #pragma unroll
    for (int off = 1; off < 16; off <<= 1)
        #pragma unroll
        for (int j = 0; j < 8; ++j) {
            rs[j] += __shfl_xor(rs[j], off, 32);
            aT[j] += __shfl_xor(aT[j], off, 32);
        }
    float ssq[8] = {};
    #pragma unroll
    for (int nt = 0; nt < 4; ++nt)
        #pragma unroll
        for (int j = 0; j < 8; ++j) ssq[j] += O[nt][j] * O[nt][j];
    #pragma unroll
    for (int off = 1; off < 16; off <<= 1)
        #pragma unroll
        for (int j = 0; j < 8; ++j) ssq[j] += __shfl_xor(ssq[j], off, 32);

    float scale[8];
    #pragma unroll
    for (int j = 0; j < 8; ++j) {
        float m = aT[j] * aT[j] - ssq[j];           // -mink of unnormalized agg
        float cl = fmaxf(m, EPSF * rs[j] * rs[j]);  // clip at EPS * rowsum^2
        scale[j] = rsqrtf(fmaxf(cl, 1e-30f));
    }
    #pragma unroll
    for (int nt = 0; nt < 4; ++nt)
        #pragma unroll
        for (int j = 0; j < 8; ++j) {
            int row  = q0 + wave * 16 + (lh << 3) + j;
            int dcol = nt * 16 + lm;
            size_t idx = ((size_t)b * L + row) * D + h * dh + dcol;
            float v = O[nt][j] * scale[j];
            unsigned short hh = f2bf(v);
            aggHi[idx] = hh;
            aggLo[idx] = f2bf(v - bf2f(hh));
        }
}

// ---------------------------------------------------------------------------
// x1 = hln(midpoint(z, x)): one wave per 512-dim row (8 rows / 256-thr block)
// ---------------------------------------------------------------------------
__global__ __launch_bounds__(256) void k_mid(
    const float* __restrict__ zS, const float* __restrict__ x,
    const float* __restrict__ g, const float* __restrict__ be,
    float* __restrict__ x1S, unsigned short* __restrict__ x1Hi,
    unsigned short* __restrict__ x1Lo, float* __restrict__ x1T,
    int rows, int D) {
    const int lane = threadIdx.x & 31, wave = threadIdx.x >> 5;
    const int row = blockIdx.x * 8 + wave;
    if (row >= rows) return;
    const float* zr = zS + (size_t)row * D;
    const float* xr = x + (size_t)row * (D + 1);

    float a[16], zz = 0.f;
    #pragma unroll
    for (int i = 0; i < 16; ++i) {
        float z = zr[lane + 32 * i];
        zz += z * z;
        a[i] = z;
    }
    zz = wreduce32(zz);
    float at_ = sqrtf(1.f + zz) + xr[0];

    float sa = 0.f;
    #pragma unroll
    for (int i = 0; i < 16; ++i) {
        a[i] += xr[1 + lane + 32 * i];
        sa += a[i] * a[i];
    }
    sa = wreduce32(sa);
    float inv = rsqrtf(fmaxf(at_ * at_ - sa, EPSF));

    float su = 0.f, sq = 0.f;
    #pragma unroll
    for (int i = 0; i < 16; ++i) {
        a[i] *= inv;
        su += a[i];
        sq += a[i] * a[i];
    }
    su = wreduce32(su);
    sq = wreduce32(sq);
    float mu = su / D, var = sq / D - mu * mu;
    float r = rsqrtf(var + EPSF);

    float yy = 0.f;
    #pragma unroll
    for (int i = 0; i < 16; ++i) {
        int d = lane + 32 * i;
        float y = (a[i] - mu) * r * g[d] + be[d];
        size_t idx = (size_t)row * D + d;
        x1S[idx] = y;
        unsigned short h = f2bf(y);
        x1Hi[idx] = h;
        x1Lo[idx] = f2bf(y - bf2f(h));
        yy += y * y;
    }
    yy = wreduce32(yy);
    if (lane == 0) x1T[row] = sqrtf(1.f + yy);
}

// ---------------------------------------------------------------------------
// out = hln(midpoint(z2, x1)), written lifted (time + 512 spatial)
// ---------------------------------------------------------------------------
__global__ __launch_bounds__(256) void k_final(
    const float* __restrict__ z2S, const float* __restrict__ x1S,
    const float* __restrict__ x1T, const float* __restrict__ g,
    const float* __restrict__ be, float* __restrict__ out,
    int rows, int D) {
    const int lane = threadIdx.x & 31, wave = threadIdx.x >> 5;
    const int row = blockIdx.x * 8 + wave;
    if (row >= rows) return;
    const float* zr = z2S + (size_t)row * D;
    const float* xr = x1S + (size_t)row * D;

    float a[16], zz = 0.f;
    #pragma unroll
    for (int i = 0; i < 16; ++i) {
        float z = zr[lane + 32 * i];
        zz += z * z;
        a[i] = z;
    }
    zz = wreduce32(zz);
    float at_ = sqrtf(1.f + zz) + x1T[row];

    float sa = 0.f;
    #pragma unroll
    for (int i = 0; i < 16; ++i) {
        a[i] += xr[lane + 32 * i];
        sa += a[i] * a[i];
    }
    sa = wreduce32(sa);
    float inv = rsqrtf(fmaxf(at_ * at_ - sa, EPSF));

    float su = 0.f, sq = 0.f;
    #pragma unroll
    for (int i = 0; i < 16; ++i) {
        a[i] *= inv;
        su += a[i];
        sq += a[i] * a[i];
    }
    su = wreduce32(su);
    sq = wreduce32(sq);
    float mu = su / D, var = sq / D - mu * mu;
    float r = rsqrtf(var + EPSF);

    float yy = 0.f;
    #pragma unroll
    for (int i = 0; i < 16; ++i) {
        int d = lane + 32 * i;
        float y = (a[i] - mu) * r * g[d] + be[d];
        out[(size_t)row * (D + 1) + 1 + d] = y;
        yy += y * y;
    }
    yy = wreduce32(yy);
    if (lane == 0) out[(size_t)row * (D + 1)] = sqrtf(1.f + yy);
}

// ---------------------------------------------------------------------------
extern "C" void kernel_launch(void* const* d_in, const int* in_sizes, int n_in,
                              void* d_out, int out_size, void* d_ws, size_t ws_size,
                              hipStream_t stream) {
    const float* x   = (const float*)d_in[0];
    const float* Wq  = (const float*)d_in[1];  const float* bq  = (const float*)d_in[2];
    const float* Wk  = (const float*)d_in[3];  const float* bk  = (const float*)d_in[4];
    const float* Wv  = (const float*)d_in[5];  const float* bv  = (const float*)d_in[6];
    const float* Wo  = (const float*)d_in[7];  const float* bo  = (const float*)d_in[8];
    const float* W1  = (const float*)d_in[9];  const float* b1  = (const float*)d_in[10];
    const float* W2  = (const float*)d_in[11]; const float* b2  = (const float*)d_in[12];
    const float* g1  = (const float*)d_in[13]; const float* be1 = (const float*)d_in[14];
    const float* g2  = (const float*)d_in[15]; const float* be2 = (const float*)d_in[16];

    constexpr int Bb = 8, L = 1024, D = 512, H = 8, dh = 64, MLP = 2048;
    constexpr int BL = Bb * L;  // 8192

    char* w = (char*)d_ws;
    auto alloc = [&](size_t bytes) -> void* {
        void* p = (void*)w;
        w += (bytes + 255) & ~(size_t)255;
        return p;
    };
    typedef unsigned short u16;
    u16* sHi  = (u16*)alloc((size_t)BL * D * 2);  u16* sLo  = (u16*)alloc((size_t)BL * D * 2);
    u16* WqHi = (u16*)alloc((size_t)D * D * 2);   u16* WqLo = (u16*)alloc((size_t)D * D * 2);
    u16* WkHi = (u16*)alloc((size_t)D * D * 2);   u16* WkLo = (u16*)alloc((size_t)D * D * 2);
    u16* WvHi = (u16*)alloc((size_t)D * D * 2);   u16* WvLo = (u16*)alloc((size_t)D * D * 2);
    u16* WoHi = (u16*)alloc((size_t)D * D * 2);   u16* WoLo = (u16*)alloc((size_t)D * D * 2);
    u16* W1Hi = (u16*)alloc((size_t)MLP * D * 2); u16* W1Lo = (u16*)alloc((size_t)MLP * D * 2);
    u16* W2Hi = (u16*)alloc((size_t)D * MLP * 2); u16* W2Lo = (u16*)alloc((size_t)D * MLP * 2);
    u16* qHi  = (u16*)alloc((size_t)BL * D * 2);  u16* qLo  = (u16*)alloc((size_t)BL * D * 2);
    u16* kHi  = (u16*)alloc((size_t)BL * D * 2);  u16* kLo  = (u16*)alloc((size_t)BL * D * 2);
    u16* vHi  = (u16*)alloc((size_t)BL * D * 2);  u16* vLo  = (u16*)alloc((size_t)BL * D * 2);
    u16* aHi  = (u16*)alloc((size_t)BL * D * 2);  u16* aLo  = (u16*)alloc((size_t)BL * D * 2);
    float* zS  = (float*)alloc((size_t)BL * D * 4);
    float* x1S = (float*)alloc((size_t)BL * D * 4);
    u16* x1Hi  = (u16*)alloc((size_t)BL * D * 2); u16* x1Lo = (u16*)alloc((size_t)BL * D * 2);
    float* x1T = (float*)alloc((size_t)BL * 4);
    u16* h1Hi  = (u16*)alloc((size_t)BL * MLP * 2);
    u16* h1Lo  = (u16*)alloc((size_t)BL * MLP * 2);
    float* z2S = (float*)alloc((size_t)BL * D * 4);

    auto blocks = [](size_t n) { return (unsigned)((n + 255) / 256); };

    // 1. splits
    k_split_x<<<blocks((size_t)BL * D), 256, 0, stream>>>(x, sHi, sLo, BL, D);
    k_split<<<blocks((size_t)D * D), 256, 0, stream>>>(Wq, WqHi, WqLo, (size_t)D * D);
    k_split<<<blocks((size_t)D * D), 256, 0, stream>>>(Wk, WkHi, WkLo, (size_t)D * D);
    k_split<<<blocks((size_t)D * D), 256, 0, stream>>>(Wv, WvHi, WvLo, (size_t)D * D);
    k_split<<<blocks((size_t)D * D), 256, 0, stream>>>(Wo, WoHi, WoLo, (size_t)D * D);
    k_split<<<blocks((size_t)MLP * D), 256, 0, stream>>>(W1, W1Hi, W1Lo, (size_t)MLP * D);
    k_split<<<blocks((size_t)D * MLP), 256, 0, stream>>>(W2, W2Hi, W2Lo, (size_t)D * MLP);

    // 2. QKV projections (WMMA)
    dim3 gD(D / BN, BL / BM), gM(MLP / BN, BL / BM);
    gemm_bf16x2<0><<<gD, 256, 0, stream>>>(sHi, sLo, WqHi, WqLo, bq, nullptr, qHi, qLo, BL, D, D);
    gemm_bf16x2<0><<<gD, 256, 0, stream>>>(sHi, sLo, WkHi, WkLo, bk, nullptr, kHi, kLo, BL, D, D);
    gemm_bf16x2<0><<<gD, 256, 0, stream>>>(sHi, sLo, WvHi, WvLo, bv, nullptr, vHi, vLo, BL, D, D);

    // 3. hyperbolic attention (WMMA)
    hattn_kernel<<<Bb * H * (L / 64), 128, 0, stream>>>(qHi, qLo, kHi, kLo, vHi, vLo,
                                                        aHi, aLo, Bb, L, H, dh);

    // 4. output projection, midpoint + LN
    gemm_bf16x2<0><<<gD, 256, 0, stream>>>(aHi, aLo, WoHi, WoLo, bo, zS, nullptr, nullptr, BL, D, D);
    k_mid<<<BL / 8, 256, 0, stream>>>(zS, x, g1, be1, x1S, x1Hi, x1Lo, x1T, BL, D);

    // 5. MLP (fused ReLU + hi/lo re-split), then second midpoint + LN -> out
    gemm_bf16x2<1><<<gM, 256, 0, stream>>>(x1Hi, x1Lo, W1Hi, W1Lo, b1, nullptr, h1Hi, h1Lo, BL, MLP, D);
    gemm_bf16x2<0><<<gD, 256, 0, stream>>>(h1Hi, h1Lo, W2Hi, W2Lo, b2, z2S, nullptr, nullptr, BL, D, MLP);
    k_final<<<BL / 8, 256, 0, stream>>>(z2S, x1S, x1T, g2, be2, (float*)d_out, BL, D);
}
